// WindowMultiHeadAttention_46377056863019
// MI455X (gfx1250) — matrix operand order repaired and checked by reading, compiled, not run on hardware
//
#include <hip/hip_runtime.h>

// ---------------------------------------------------------------------------
// Types for CDNA5 WMMA (wave32): V_WMMA_F32_16X16X32_BF16
// ---------------------------------------------------------------------------
typedef __attribute__((ext_vector_type(8)))  float   v8f;
typedef __attribute__((ext_vector_type(4)))  float   v4f;
typedef __attribute__((ext_vector_type(16))) __bf16  v16bf;
typedef __attribute__((ext_vector_type(8)))  __bf16  v8bf;

__device__ __forceinline__ v16bf bcat(v8bf lo, v8bf hi) {
  return __builtin_shufflevector(lo, hi, 0,1,2,3,4,5,6,7,8,9,10,11,12,13,14,15);
}

__device__ __forceinline__ v8bf cvt8f(const float* p) {
  v4f a = *(const v4f*)p;
  v4f b = *(const v4f*)(p + 4);
  v8bf r;
  r[0] = (__bf16)a[0]; r[1] = (__bf16)a[1]; r[2] = (__bf16)a[2]; r[3] = (__bf16)a[3];
  r[4] = (__bf16)b[0]; r[5] = (__bf16)b[1]; r[6] = (__bf16)b[2]; r[7] = (__bf16)b[3];
  return r;
}

__device__ __forceinline__ v8f wmma_bf(v16bf a, v16bf b, v8f c) {
  // D = A(16x32 bf16) * B(32x16 bf16) + C(16x16 f32)
  return __builtin_amdgcn_wmma_f32_16x16x32_bf16(false, a, false, b, (short)0, c,
                                                 false, false);
}

// reductions confined to 16-lane half-waves (xor masks 1..8 never cross 16)
__device__ __forceinline__ float red_max16(float v) {
  v = fmaxf(v, __shfl_xor(v, 1, 32));
  v = fmaxf(v, __shfl_xor(v, 2, 32));
  v = fmaxf(v, __shfl_xor(v, 4, 32));
  v = fmaxf(v, __shfl_xor(v, 8, 32));
  return v;
}
__device__ __forceinline__ float red_sum16(float v) {
  v += __shfl_xor(v, 1, 32);
  v += __shfl_xor(v, 2, 32);
  v += __shfl_xor(v, 4, 32);
  v += __shfl_xor(v, 8, 32);
  return v;
}

#define B_   256
#define C_   256
#define T_   256
#define NH_  8
#define HD_  32
#define MH_  256
#define TPAD 40    // LDS row pitch (halves) for x^T panels: conflict-free B128 reads

// ---------------------------------------------------------------------------
// Kernel 1: relative-position MLP bias  (batch-invariant; computed once)
// bias[h][q][k],  65536 threads, one (q,k) pair each; weights are uniform
// across lanes -> scalar loads.
// ---------------------------------------------------------------------------
__global__ __launch_bounds__(256)
void bias_mlp_kernel(const float* __restrict__ mw1, const float* __restrict__ mb1,
                     const float* __restrict__ mw2, const float* __restrict__ mb2,
                     float* __restrict__ bias) {
  int p = blockIdx.x * 256 + threadIdx.x;       // 0 .. 65535
  int q = p >> 8, k = p & 255;
  float dy = (float)(q >> 4) - (float)(k >> 4);
  float dx = (float)(q & 15) - (float)(k & 15);
  float r0 = copysignf(log1pf(fabsf(dy)), dy);
  float r1 = copysignf(log1pf(fabsf(dx)), dx);
  float acc[NH_];
#pragma unroll
  for (int h = 0; h < NH_; ++h) acc[h] = mb2[h];
  for (int m = 0; m < MH_; ++m) {
    float hm = fmaxf(mw1[2 * m] * r0 + mw1[2 * m + 1] * r1 + mb1[m], 0.0f);
#pragma unroll
    for (int h = 0; h < NH_; ++h) acc[h] += mw2[h * MH_ + m] * hm;
  }
#pragma unroll
  for (int h = 0; h < NH_; ++h) bias[(size_t)h * 65536 + p] = acc[h];
}

// ---------------------------------------------------------------------------
// stage a [32 c][256 t] fp32 panel of x as x^T bf16 into LDS [256 t][TPAD]
// ---------------------------------------------------------------------------
__device__ __forceinline__ void stage_xT(__bf16* xT, const float* xb, int kk, int tid) {
  const int cl = tid >> 3;            // 0..31 local c row
  const int t0 = (tid & 7) * 32;      // t window
  const float* src = xb + (size_t)(kk * 32 + cl) * T_ + t0;
#pragma unroll
  for (int i = 0; i < 32; i += 4) {
    v4f v = *(const v4f*)(src + i);
#pragma unroll
    for (int j = 0; j < 4; ++j) xT[(t0 + i + j) * TPAD + cl] = (__bf16)v[j];
  }
}

// ---------------------------------------------------------------------------
// Kernel 2: QKV GEMM.  qkv^T per window: [768 j][256 t] = w_qkv * x[b]
//  part 0..3: q/k in M=t orientation  -> qbf/kbf[b][h][t][d] + row norms
//  part 4..5: v   in M=j orientation  -> vT [b][h][d][t]   (AV B-layout)
// ---------------------------------------------------------------------------
__global__ __launch_bounds__(256)
void qkv_gemm_kernel(const float* __restrict__ x, const float* __restrict__ w_qkv,
                     const float* __restrict__ b_qkv,
                     __bf16* __restrict__ qbf, __bf16* __restrict__ kbf,
                     __bf16* __restrict__ vT,
                     float* __restrict__ ssq_q, float* __restrict__ ssq_k) {
  __shared__ __bf16 xT[T_ * TPAD];
  const int b = blockIdx.x, part = blockIdx.y;
  const int tid = threadIdx.x, wid = tid >> 5, lane = tid & 31;
  const int half = lane >> 4, l16 = lane & 15;
  const float* xb = x + (size_t)b * C_ * T_;

  if (part < 4) {                      // ---- Q and K: C-tile rows = t ----
    const int region = part >> 1;      // 0=q, 1=k
    const int tb = (part & 1) * 128 + wid * 16;
    const int jbase = region * 256;
    v8f acc[16];
#pragma unroll
    for (int n = 0; n < 16; ++n) acc[n] = (v8f){};
    for (int kk = 0; kk < 8; ++kk) {
      __syncthreads();
      stage_xT(xT, xb, kk, tid);
      __syncthreads();
      // A (16 t x 32 c) from LDS x^T
      const __bf16* ap = &xT[(tb + l16) * TPAD + half * 16];
      v16bf A = bcat(*(const v8bf*)ap, *(const v8bf*)(ap + 8));
#pragma unroll
      for (int n = 0; n < 16; ++n) {   // B columns = w_qkv rows (fp32 -> bf16)
        const float* wrow =
            w_qkv + (size_t)(jbase + n * 16 + l16) * C_ + kk * 32 + half * 16;
        if (kk < 7) __builtin_prefetch(wrow + 32, 0, 1);
        acc[n] = wmma_bf(A, bcat(cvt8f(wrow), cvt8f(wrow + 8)), acc[n]);
      }
    }
    // epilogue: + b_qkv, row sum-of-squares (norms), bf16 store
    float bq[16];
#pragma unroll
    for (int n = 0; n < 16; ++n) bq[n] = b_qkv[jbase + n * 16 + l16];
#pragma unroll
    for (int n = 0; n < 16; ++n)
#pragma unroll
      for (int r = 0; r < 8; ++r) acc[n][r] += bq[n];
    __bf16* dst = (region == 0) ? qbf : kbf;
    float* ssq = (region == 0) ? ssq_q : ssq_k;
#pragma unroll
    for (int h = 0; h < 8; ++h)
#pragma unroll
      for (int r = 0; r < 8; ++r) {
        float v0 = acc[2 * h][r], v1 = acc[2 * h + 1][r];
        float ss = red_sum16(v0 * v0 + v1 * v1);   // over d (2 tiles x 16 lanes)
        if (l16 == 0) ssq[(((b * 8 + h) << 8)) + tb + r + 8 * half] = ss;
      }
#pragma unroll
    for (int n = 0; n < 16; ++n) {
      int h = n >> 1, d = (n & 1) * 16 + l16;
#pragma unroll
      for (int r = 0; r < 8; ++r) {
        int t = tb + r + 8 * half;
        dst[((((size_t)b * 8 + h) << 8) + t) * HD_ + d] = (__bf16)acc[n][r];
      }
    }
  } else {                             // ---- V: C-tile rows = j (=> vT[d][t]) ----
    const int jb = 512 + (part - 4) * 128 + wid * 16;
    v8f acc[16];
#pragma unroll
    for (int n = 0; n < 16; ++n) acc[n] = (v8f){};
    for (int kk = 0; kk < 8; ++kk) {
      __syncthreads();
      stage_xT(xT, xb, kk, tid);
      __syncthreads();
      const float* wrow = w_qkv + (size_t)(jb + l16) * C_ + kk * 32 + half * 8;
      v16bf A = bcat(cvt8f(wrow), cvt8f(wrow + 16));
#pragma unroll
      for (int n = 0; n < 16; ++n) {   // B columns = x^T rows from LDS
        const __bf16* bp = &xT[(n * 16 + l16) * TPAD + half * 8];
        acc[n] = wmma_bf(A, bcat(*(const v8bf*)bp, *(const v8bf*)(bp + 16)), acc[n]);
      }
    }
#pragma unroll
    for (int r = 0; r < 8; ++r) {
      int j = jb + r + 8 * half;
      float bv = b_qkv[j];
      int h = (j - 512) >> 5, d = (j - 512) & 31;
#pragma unroll
      for (int n = 0; n < 16; ++n) {
        int t = n * 16 + l16;
        vT[((((size_t)b * 8 + h) * HD_ + d) << 8) + t] = (__bf16)(acc[n][r] + bv);
      }
    }
  }
}

// ---------------------------------------------------------------------------
// Kernel 3: fused cosine attention.  1 wave = 16-row q strip of one (b,h).
// 128 threads / 4 waves -> 64 q rows per WG; grid = 256*8*4.
// ---------------------------------------------------------------------------
#define PPITCH 264  // probs LDS pitch (halves): conflict-free A-tile reads

__global__ __launch_bounds__(128)
void attn_kernel(const __bf16* __restrict__ qbf, const __bf16* __restrict__ kbf,
                 const __bf16* __restrict__ vT,
                 const float* __restrict__ ssq_q, const float* __restrict__ ssq_k,
                 const float* __restrict__ bias, const float* __restrict__ tau,
                 __bf16* __restrict__ o) {
  __shared__ __bf16 probs[4 * 16 * PPITCH];
  const int bid = blockIdx.x;
  const int b = bid >> 5, h = (bid >> 2) & 7, qblk = bid & 3;
  const int tid = threadIdx.x, wid = tid >> 5, lane = tid & 31;
  const int half = lane >> 4, l16 = lane & 15;
  const int qb = qblk * 64 + wid * 16;

  const __bf16* qp = qbf + ((((size_t)b * 8 + h) << 8) * HD_);
  const __bf16* kp = kbf + ((((size_t)b * 8 + h) << 8) * HD_);
  const __bf16* vp = vT + ((((size_t)b * 8 + h) * HD_) << 8);
  const float* sq = ssq_q + ((b * 8 + h) << 8);
  const float* sk = ssq_k + ((b * 8 + h) << 8);
  const float* bi = bias + ((size_t)h << 16);
  const float itau = 1.0f / fmaxf(tau[h], 0.01f);

  // A_q : 16 q rows x 32 d
  const __bf16* qrow = qp + (size_t)(qb + l16) * HD_ + half * 8;
  v16bf Aq = bcat(*(const v8bf*)qrow, *(const v8bf*)(qrow + 16));
  float qn[8];
#pragma unroll
  for (int r = 0; r < 8; ++r) qn[r] = sq[qb + r + 8 * half];

  // scores strip 16 x 256 : QK^T (K dim = full hd = 32 in one wmma each)
  v8f s[16];
  const v8f zro = (v8f){};
#pragma unroll
  for (int n = 0; n < 16; ++n) {
    const __bf16* krow = kp + (size_t)(n * 16 + l16) * HD_ + half * 8;
    v16bf Bk = bcat(*(const v8bf*)krow, *(const v8bf*)(krow + 16));
    s[n] = wmma_bf(Aq, Bk, zro);
  }
  // cosine normalize, tau, bias
#pragma unroll
  for (int n = 0; n < 16; ++n) {
    float kn = sk[n * 16 + l16];
#pragma unroll
    for (int r = 0; r < 8; ++r) {
      float denom = fmaxf(sqrtf(qn[r] * kn), 1e-6f);
      float bv = bi[(size_t)(qb + r + 8 * half) * T_ + n * 16 + l16];
      s[n][r] = (s[n][r] / denom) * itau + bv;
    }
  }
  // softmax over k (row spans 16 lanes x 16 tiles)
#pragma unroll
  for (int r = 0; r < 8; ++r) {
    float m = -3.4e38f;
#pragma unroll
    for (int n = 0; n < 16; ++n) m = fmaxf(m, s[n][r]);
    m = red_max16(m);
    float sum = 0.0f;
#pragma unroll
    for (int n = 0; n < 16; ++n) { float e = __expf(s[n][r] - m); s[n][r] = e; sum += e; }
    sum = red_sum16(sum);
    float inv = 1.0f / sum;
#pragma unroll
    for (int n = 0; n < 16; ++n) s[n][r] *= inv;
  }
  // probs -> LDS (C-layout scatter, per-wave strip), then AV with wmma
  __bf16* pb = probs + wid * 16 * PPITCH;
#pragma unroll
  for (int n = 0; n < 16; ++n)
#pragma unroll
    for (int r = 0; r < 8; ++r)
      pb[(r + 8 * half) * PPITCH + n * 16 + l16] = (__bf16)s[n][r];

  v8f o0 = (v8f){}, o1 = (v8f){};
#pragma unroll
  for (int j2 = 0; j2 < 8; ++j2) {
    const __bf16* prow = pb + l16 * PPITCH + j2 * 32 + half * 16;
    v16bf Ap = bcat(*(const v8bf*)prow, *(const v8bf*)(prow + 8));
    const __bf16* vr0 = vp + (size_t)(l16) * T_ + j2 * 32 + half * 16;
    o0 = wmma_bf(Ap, bcat(*(const v8bf*)vr0, *(const v8bf*)(vr0 + 8)), o0);
    const __bf16* vr1 = vp + (size_t)(16 + l16) * T_ + j2 * 32 + half * 16;
    o1 = wmma_bf(Ap, bcat(*(const v8bf*)vr1, *(const v8bf*)(vr1 + 8)), o1);
  }
  // store o[b][t][c] bf16 (rows t, channels c = h*32 + d)
#pragma unroll
  for (int r = 0; r < 8; ++r) {
    int t = qb + r + 8 * half;
    __bf16* op = o + (((size_t)b << 8) + t) * C_ + h * HD_;
    op[l16] = (__bf16)o0[r];
    op[16 + l16] = (__bf16)o1[r];
  }
}

// ---------------------------------------------------------------------------
// Kernel 4: output projection, computed as y^T = w_proj * o^T so that the
// [B,C,H,W] stores are coalesced.  1 wave = 16 output channels x 256 t.
// ---------------------------------------------------------------------------
__global__ __launch_bounds__(256)
void proj_kernel(const __bf16* __restrict__ o, const float* __restrict__ w_proj,
                 const float* __restrict__ b_proj, float* __restrict__ y) {
  const int b = blockIdx.x;
  const int tid = threadIdx.x, wid = tid >> 5, lane = tid & 31;
  const int half = lane >> 4, l16 = lane & 15;
  const int cob = blockIdx.y * 128 + wid * 16;
  const __bf16* ob = o + ((size_t)b << 16);   // [256 t][256 c] bf16
  v8f acc[16];
#pragma unroll
  for (int n = 0; n < 16; ++n) acc[n] = (v8f){};
  for (int kk = 0; kk < 8; ++kk) {
    const float* wrow = w_proj + (size_t)(cob + l16) * C_ + kk * 32 + half * 16;
    v16bf A = bcat(cvt8f(wrow), cvt8f(wrow + 8));
#pragma unroll
    for (int n = 0; n < 16; ++n) {             // B columns = o rows
      const __bf16* orow = ob + (size_t)(n * 16 + l16) * C_ + kk * 32 + half * 16;
      acc[n] = wmma_bf(A, bcat(*(const v8bf*)orow, *(const v8bf*)(orow + 8)), acc[n]);
    }
  }
#pragma unroll
  for (int r = 0; r < 8; ++r) {
    int co = cob + r + 8 * half;
    float bp = b_proj[co];
    float* yrow = y + (((size_t)b << 8) + co) * T_;
#pragma unroll
    for (int n = 0; n < 16; ++n) yrow[n * 16 + l16] = acc[n][r] + bp;
  }
}

// ---------------------------------------------------------------------------
extern "C" void kernel_launch(void* const* d_in, const int* in_sizes, int n_in,
                              void* d_out, int out_size, void* d_ws, size_t ws_size,
                              hipStream_t stream) {
  const float* x      = (const float*)d_in[0];
  const float* w_qkv  = (const float*)d_in[1];
  const float* b_qkv  = (const float*)d_in[2];
  const float* w_proj = (const float*)d_in[3];
  const float* b_proj = (const float*)d_in[4];
  const float* mw1    = (const float*)d_in[5];
  const float* mb1    = (const float*)d_in[6];
  const float* mw2    = (const float*)d_in[7];
  const float* mb2    = (const float*)d_in[8];
  const float* tau    = (const float*)d_in[9];
  float* y = (float*)d_out;

  char* ws = (char*)d_ws;
  size_t off = 0;
  float*  bias  = (float*)(ws + off);  off += (size_t)8 * 65536 * 4;       // 2 MB
  __bf16* qbf   = (__bf16*)(ws + off); off += (size_t)256 * 8 * 256 * 32 * 2;
  __bf16* kbf   = (__bf16*)(ws + off); off += (size_t)256 * 8 * 256 * 32 * 2;
  __bf16* vT    = (__bf16*)(ws + off); off += (size_t)256 * 8 * 256 * 32 * 2;
  __bf16* oattn = (__bf16*)(ws + off); off += (size_t)256 * 256 * 256 * 2;
  float*  ssq_q = (float*)(ws + off);  off += (size_t)256 * 8 * 256 * 4;
  float*  ssq_k = (float*)(ws + off);  off += (size_t)256 * 8 * 256 * 4;

  bias_mlp_kernel<<<256, 256, 0, stream>>>(mw1, mb1, mw2, mb2, bias);
  qkv_gemm_kernel<<<dim3(256, 6), 256, 0, stream>>>(x, w_qkv, b_qkv, qbf, kbf, vT,
                                                    ssq_q, ssq_k);
  attn_kernel<<<8192, 128, 0, stream>>>(qbf, kbf, vT, ssq_q, ssq_k, bias, tau, oattn);
  proj_kernel<<<dim3(256, 2), 256, 0, stream>>>(oattn, w_proj, b_proj, y);
}
